// ARSG_17042430231273
// MI455X (gfx1250) — matrix operand order repaired and checked by reading, not compiled, not run on hardware
//
#include <hip/hip_runtime.h>
#include <hip/hip_bf16.h>

#define T_DIM 1024
#define B_DIM 32
#define DF 512
#define DH 512
#define DS_DIM 1024
#define DW 512
#define AEXT 1088   // [16 zeros | 1024 a values | 48 zeros], bf16
#define RPAD 2080   // FUpadT cols: [512 zeros | 1024 FU rows | 544 zeros], bf16

typedef __attribute__((ext_vector_type(16))) __bf16 v16bf;
typedef __attribute__((ext_vector_type(8)))  float  v8f;

__device__ __forceinline__ unsigned short f2bf(float f) {
  union { float f; unsigned u; } c; c.f = f;
  unsigned u = c.u;
  return (unsigned short)((u + 0x7FFFu + ((u >> 16) & 1u)) >> 16); // RNE
}
__device__ __forceinline__ unsigned pack2(float a, float b) {
  return (unsigned)f2bf(a) | ((unsigned)f2bf(b) << 16);
}

union AF { v16bf v; unsigned u[8]; uint4 q[2]; };

// Load a 32x16 bf16 B-fragment: 8 contiguous u32 per lane, 16B aligned.
// base16: bf16 element index of (row n, k0); caller adds 16*half (contiguous k = 16*half + s, as in gatherA).
__device__ __forceinline__ void loadB(const unsigned short* __restrict__ p, AF& f) {
  const uint4* q = (const uint4*)p;
  f.q[0] = q[0];
  f.q[1] = q[1];
}

// Gather a 16x32 bf16 A-fragment from an LDS tile (row-major 16x32 bf16 = 16x16 u32).
__device__ __forceinline__ void gatherA(const unsigned* __restrict__ row, int half, AF& f) {
  const uint4* q = (const uint4*)(row + half * 8);
  f.q[0] = q[0];   // K = 0..7   (+16*half)
  f.q[1] = q[1];   // K = 8..15  (+16*half)
}

// ---------------- small prep kernels ----------------

__global__ void k_sWb(const float* __restrict__ s_prev, const float* __restrict__ Ww,
                      const float* __restrict__ Vb, float* __restrict__ sWb) {
  int idx = blockIdx.x * blockDim.x + threadIdx.x;   // 32*512
  int b = idx >> 9, w = idx & 511;
  const float* sp = s_prev + b * DS_DIM;
  float acc = 0.f;
  for (int s = 0; s < DS_DIM; ++s) acc += sp[s] * Ww[s * DW + w];
  sWb[idx] = acc + Vb[w];
}

__global__ void k_convert(const float* __restrict__ a_prev, const float* __restrict__ Vw,
                          const float* __restrict__ Uw,
                          unsigned short* __restrict__ aext,
                          unsigned short* __restrict__ VwT,
                          unsigned short* __restrict__ UwT,
                          unsigned short* __restrict__ FUp) {
  const int NA = B_DIM * AEXT;          // 34816
  const int NV = DW * DH;               // 262144
  const int NU = DW * DF;               // 262144
  const int NF = DW * RPAD;             // 1064960
  for (int i = blockIdx.x * blockDim.x + threadIdx.x; i < NA + NV + NU + NF;
       i += gridDim.x * blockDim.x) {
    if (i < NA) {
      int b = i / AEXT, j = i - b * AEXT;
      int s = j - 16;
      aext[i] = (s >= 0 && s < T_DIM) ? f2bf(a_prev[b * T_DIM + s]) : (unsigned short)0;
    } else if (i < NA + NV) {
      int t = i - NA; int w = t >> 9; int k = t & 511;
      VwT[w * DH + k] = f2bf(Vw[k * DW + w]);     // transpose: K-pairs contiguous
    } else if (i < NA + NV + NU) {
      int t = i - NA - NV; int w = t >> 9; int k = t & 511;
      UwT[w * DF + k] = f2bf(Uw[k * DW + w]);
    } else {
      FUp[i - NA - NV - NU] = 0;                   // zero pads (interior overwritten by k_fu)
    }
  }
}

// ---------------- FU = F @ Uw  (WMMA, bf16 -> f32), stored transposed+padded ----------------

__global__ void __launch_bounds__(256) k_fu(const float* __restrict__ F,
                                            const unsigned short* __restrict__ UwT,
                                            unsigned short* __restrict__ FUp) {
  __shared__ alignas(16) unsigned Al[2][16][16];   // double-buffered 16x32 bf16 A tile
  const int tid  = threadIdx.x;
  const int lane = tid & 31, wv = tid >> 5;
  const int m = lane & 15, half = lane >> 4;
  const int t0 = blockIdx.x * 16;
  const int sm = (tid * 2) >> 5, sk = (tid * 2) & 31;   // staging coords
  v8f c[4] = {};

  // prologue stage kb=0
  {
    const float* src = F + (t0 + sm) * DF + sk;
    Al[0][sm][sk >> 1] = pack2(src[0], src[1]);
  }
  for (int kb = 0; kb < DF / 32; ++kb) {
    __syncthreads();
    if (kb + 1 < DF / 32) {
      const float* src = F + (t0 + sm) * DF + (kb + 1) * 32 + sk;
      Al[(kb + 1) & 1][sm][sk >> 1] = pack2(src[0], src[1]);
    }
    AF a; gatherA(Al[kb & 1][m], half, a);
#pragma unroll
    for (int qd = 0; qd < 4; ++qd) {
      int n = wv * 64 + qd * 16 + m;
      AF bfr; loadB(UwT + n * DF + kb * 32 + 16 * half, bfr);
      c[qd] = __builtin_amdgcn_wmma_f32_16x16x32_bf16(false, a.v, false, bfr.v,
                                                      (short)0, c[qd], false, false);
    }
  }
  // FUpadT[w][512 + t] = bf16(FU[t][w])
#pragma unroll
  for (int qd = 0; qd < 4; ++qd) {
    int n = wv * 64 + qd * 16 + m;
#pragma unroll
    for (int v = 0; v < 8; ++v) {
      int M = v + 8 * half;
      FUp[n * RPAD + 512 + t0 + M] = f2bf(c[qd][v]);
    }
  }
}

// ---------------- fused: y = hT@Vw + Toeplitz(a)@FU + sWb; e = tanh(y).ww ----------------

__global__ void __launch_bounds__(256) k_fused(const float* __restrict__ h,
                                               const unsigned short* __restrict__ VwT,
                                               const unsigned short* __restrict__ aext,
                                               const unsigned short* __restrict__ FUp,
                                               const float* __restrict__ sWb,
                                               const float* __restrict__ ww,
                                               float* __restrict__ e) {
  __shared__ alignas(16) unsigned Al[2][16][16];
  __shared__ float red[16];
  const int tid  = threadIdx.x;
  const int lane = tid & 31, wv = tid >> 5;
  const int m = lane & 15, half = lane >> 4;
  const int t0 = blockIdx.x * 16;
  const int b  = blockIdx.y;
  const int sm = (tid * 2) >> 5, sk = (tid * 2) & 31;
  v8f c[4] = {};

  // ---- part 1: hT[b, t0..+16, :] @ Vw ----
  const float* hbase = h + (size_t)t0 * (B_DIM * DH) + b * DH;
  {
    const float* src = hbase + (size_t)sm * (B_DIM * DH) + sk;
    Al[0][sm][sk >> 1] = pack2(src[0], src[1]);
  }
  for (int kb = 0; kb < DH / 32; ++kb) {
    __syncthreads();
    if (kb + 1 < DH / 32) {
      const float* src = hbase + (size_t)sm * (B_DIM * DH) + (kb + 1) * 32 + sk;
      Al[(kb + 1) & 1][sm][sk >> 1] = pack2(src[0], src[1]);
    }
    AF a; gatherA(Al[kb & 1][m], half, a);
#pragma unroll
    for (int qd = 0; qd < 4; ++qd) {
      int n = wv * 64 + qd * 16 + m;
      AF bfr; loadB(VwT + n * DH + kb * 32 + 16 * half, bfr);
      c[qd] = __builtin_amdgcn_wmma_f32_16x16x32_bf16(false, a.v, false, bfr.v,
                                                      (short)0, c[qd], false, false);
    }
  }

  // ---- part 2: g[t,w] = sum_j a[b,j] * FUpad[t+j, w] via Toeplitz A tiles ----
  const unsigned short* arow = aext + b * AEXT;
  { // stage kk=0: A2[tt][ki] = a_ext[16 + ki - tt]
    int j = 16 + sk - sm;
    Al[0][sm][sk >> 1] = (unsigned)arow[j] | ((unsigned)arow[j + 1] << 16);
  }
  for (int kk = 0; kk <= 32; ++kk) {
    __syncthreads();
    if (kk + 1 <= 32) {
      int j = 16 + (kk + 1) * 32 + sk - sm;                 // in [1, 1071]
      Al[(kk + 1) & 1][sm][sk >> 1] = (unsigned)arow[j] | ((unsigned)arow[j + 1] << 16);
    }
    AF a; gatherA(Al[kk & 1][m], half, a);
#pragma unroll
    for (int qd = 0; qd < 4; ++qd) {
      int n = wv * 64 + qd * 16 + m;
      AF bfr; loadB(FUp + n * RPAD + t0 + kk * 32 + 16 * half, bfr);  // cols < 2080
      c[qd] = __builtin_amdgcn_wmma_f32_16x16x32_bf16(false, a.v, false, bfr.v,
                                                      (short)0, c[qd], false, false);
    }
  }

  // ---- epilogue: +sWb, tanh, dot with ww, reduce over 512 w ----
  if (tid < 16) red[tid] = 0.f;
  __syncthreads();
  float sm8[8] = {0.f, 0.f, 0.f, 0.f, 0.f, 0.f, 0.f, 0.f};
#pragma unroll
  for (int qd = 0; qd < 4; ++qd) {
    int n = wv * 64 + qd * 16 + m;
    float sw = sWb[b * DW + n];
    float wv2 = ww[n];
#pragma unroll
    for (int v = 0; v < 8; ++v) sm8[v] += tanhf(c[qd][v] + sw) * wv2;
  }
#pragma unroll
  for (int v = 0; v < 8; ++v) atomicAdd(&red[v + 8 * half], sm8[v]);  // ds_add_f32
  __syncthreads();
  if (tid < 16) e[b * T_DIM + t0 + tid] = red[tid];
}

// ---------------- t=0 row has no conv contribution in the reference ----------------

__global__ void __launch_bounds__(256) k_fix0(const float* __restrict__ h,
                                              const float* __restrict__ Vw,
                                              const float* __restrict__ sWb,
                                              const float* __restrict__ ww,
                                              float* __restrict__ e) {
  __shared__ float red[256];
  int b = blockIdx.x, tid = threadIdx.x;
  const float* h0 = h + b * DH;                             // h[t=0, b, :]
  float s = 0.f;
  for (int wbase = 0; wbase < DW; wbase += 256) {
    int w = wbase + tid;
    float acc = sWb[b * DW + w];
    for (int k = 0; k < DH; ++k) acc += h0[k] * Vw[k * DW + w];
    s += tanhf(acc) * ww[w];
  }
  red[tid] = s; __syncthreads();
  for (int off = 128; off > 0; off >>= 1) {
    if (tid < off) red[tid] += red[tid + off];
    __syncthreads();
  }
  if (tid == 0) e[b * T_DIM] = red[0];
}

// ---------------- softmax over T per batch ----------------

__global__ void __launch_bounds__(256) k_softmax(const float* __restrict__ e,
                                                 const int* __restrict__ beta_p,
                                                 float* __restrict__ out) {
  __shared__ float red[256];
  int b = blockIdx.x, tid = threadIdx.x;
  float beta = (float)beta_p[0];
  const float* er = e + b * T_DIM;
  float mx = -3.4e38f;
  for (int t = tid; t < T_DIM; t += 256) mx = fmaxf(mx, beta * er[t]);
  red[tid] = mx; __syncthreads();
  for (int off = 128; off > 0; off >>= 1) {
    if (tid < off) red[tid] = fmaxf(red[tid], red[tid + off]);
    __syncthreads();
  }
  mx = red[0]; __syncthreads();
  float sum = 0.f;
  for (int t = tid; t < T_DIM; t += 256) sum += __expf(beta * er[t] - mx);
  red[tid] = sum; __syncthreads();
  for (int off = 128; off > 0; off >>= 1) {
    if (tid < off) red[tid] += red[tid + off];
    __syncthreads();
  }
  float inv = 1.0f / red[0];
  for (int t = tid; t < T_DIM; t += 256)
    out[b * T_DIM + t] = __expf(beta * er[t] - mx) * inv;
}

// ---------------- launcher ----------------

extern "C" void kernel_launch(void* const* d_in, const int* in_sizes, int n_in,
                              void* d_out, int out_size, void* d_ws, size_t ws_size,
                              hipStream_t stream) {
  (void)in_sizes; (void)n_in; (void)out_size; (void)ws_size;
  const float* F      = (const float*)d_in[0];
  const float* a_prev = (const float*)d_in[1];
  const float* s_prev = (const float*)d_in[2];
  const float* h      = (const float*)d_in[3];
  const float* Ww     = (const float*)d_in[4];
  const float* Vw     = (const float*)d_in[5];
  const float* Vb     = (const float*)d_in[6];
  const float* Uw     = (const float*)d_in[7];
  const float* ww     = (const float*)d_in[8];
  const int*   beta   = (const int*)d_in[9];

  char* ws = (char*)d_ws;
  float*          sWb  = (float*)(ws + 0);                  //  64 KB
  unsigned short* aext = (unsigned short*)(ws + 65536);     //  68 KB
  unsigned short* VwT  = (unsigned short*)(ws + 135168);    // 512 KB
  unsigned short* UwT  = (unsigned short*)(ws + 659456);    // 512 KB
  unsigned short* FUp  = (unsigned short*)(ws + 1183744);   // 2.03 MB
  float*          e    = (float*)(ws + 3313664);            // 128 KB  (total ~3.45 MB)

  k_sWb    <<<64, 256, 0, stream>>>(s_prev, Ww, Vb, sWb);
  k_convert<<<512, 256, 0, stream>>>(a_prev, Vw, Uw, aext, VwT, UwT, FUp);
  k_fu     <<<64, 256, 0, stream>>>(F, UwT, FUp);
  k_fused  <<<dim3(64, 32), 256, 0, stream>>>(h, VwT, aext, FUp, sWb, ww, e);
  k_fix0   <<<32, 256, 0, stream>>>(h, Vw, sWb, ww, e);
  k_softmax<<<32, 256, 0, stream>>>(e, beta, (float*)d_out);
}
